// Model_59442347376895
// MI455X (gfx1250) — matrix operand order repaired and checked by reading, not compiled, not run on hardware
//
#include <hip/hip_runtime.h>

typedef __attribute__((ext_vector_type(16))) _Float16 v16h;
typedef __attribute__((ext_vector_type(8)))  float    v8f;
typedef __attribute__((ext_vector_type(4)))  float    f4;

#define NC   11989
#define DD   200     // feature dim
#define BB   8192    // batch
#define SS   200     // samples per row
#define DPAD 224     // 7 * 32 (K padded)

// One block per b: 13 waves x 16 samples = 208 >= 200 samples.
// Each wave computes 16 dot products via v_wmma_f32_16x16x32_f16
// (B-matrix = ew chunk broadcast to all 16 columns -> matvec on the matrix pipe).
__global__ __launch_bounds__(416)
void gather_dot_sigmoid(const float* __restrict__ cpt,   // [NC, D]
                        const float* __restrict__ xg2,   // [B, D]
                        const float* __restrict__ w,     // [D, 1]
                        const int*   __restrict__ sidx,  // [B, S]
                        float*       __restrict__ out)   // [B, S]
{
    __shared__ alignas(16) float s_ew[DPAD];

    const int b    = blockIdx.x;
    const int tid  = threadIdx.x;
    const int wave = tid >> 5;
    const int lane = tid & 31;
    const int half = lane >> 4;      // 0: lanes 0-15, 1: lanes 16-31
    const int m    = lane & 15;      // row within 16-sample group

    // ew[d] = x[b,d]^2 * clip(w[d],0,1); zero-padded to 224 so tail B-chunk is 0.
    if (tid < DPAD) {
        float v = 0.0f;
        if (tid < DD) {
            float wv = fminf(fmaxf(w[tid], 0.0f), 1.0f);
            float xv = xg2[b * DD + tid];
            v = xv * xv * wv;
        }
        s_ew[tid] = v;
    }
    __syncthreads();

    const int s_base = wave * 16;
    const int s      = s_base + m;
    const int s_cl   = (s < SS) ? s : (SS - 1);          // clamp dup rows; store guarded
    const int idx    = sidx[b * SS + s_cl];
    const float* row = cpt + (long)idx * DD;

    __builtin_prefetch(row, 0, 0);   // global_prefetch_b8: warm the 800B gathered row

    const int klo = half * 8;        // 16-bit A layout: lanes16-31 hold K=8..15 / 24..31
    v8f acc = {0.f, 0.f, 0.f, 0.f, 0.f, 0.f, 0.f, 0.f};

    #pragma unroll
    for (int ci = 0; ci < 6; ++ci) {  // full K-chunks, max d = 160+31 = 191 < 200
        const int d0 = ci * 32;
        // A per-lane: two contiguous 8-float runs: [d0+klo, +8) and [d0+klo+16, +24)
        const f4 p0 = *(const f4*)(row + d0 + klo);
        const f4 p1 = *(const f4*)(row + d0 + klo + 4);
        const f4 p2 = *(const f4*)(row + d0 + klo + 16);
        const f4 p3 = *(const f4*)(row + d0 + klo + 20);
        v16h a;
        #pragma unroll
        for (int j = 0; j < 4; ++j) {
            a[j]      = (_Float16)fminf(fmaxf(p0[j], 0.f), 1.f);  // clip(relu,0,1)==clip
            a[4 + j]  = (_Float16)fminf(fmaxf(p1[j], 0.f), 1.f);
            a[8 + j]  = (_Float16)fminf(fmaxf(p2[j], 0.f), 1.f);
            a[12 + j] = (_Float16)fminf(fmaxf(p3[j], 0.f), 1.f);
        }
        // B: all 16 columns identical = ew chunk; lane half selects K 0-7,16-23 vs 8-15,24-31
        const float* ewp = s_ew + d0 + half * 8;
        const f4 q0 = *(const f4*)(ewp);
        const f4 q1 = *(const f4*)(ewp + 4);
        const f4 q2 = *(const f4*)(ewp + 16);
        const f4 q3 = *(const f4*)(ewp + 20);
        v16h bb;
        #pragma unroll
        for (int j = 0; j < 4; ++j) {
            bb[j]      = (_Float16)q0[j];
            bb[4 + j]  = (_Float16)q1[j];
            bb[8 + j]  = (_Float16)q2[j];
            bb[12 + j] = (_Float16)q3[j];
        }
        acc = __builtin_amdgcn_wmma_f32_16x16x32_f16(false, a, false, bb,
                                                     (short)0, acc, false, false);
    }

    { // tail chunk d0=192: guard gathered-row reads (d >= 200 would run off the table)
        const int d0 = 192;
        v16h a;
        #pragma unroll
        for (int j = 0; j < 16; ++j) {
            const int K = (j < 8) ? (klo + j) : (klo + 8 + j);
            const int d = d0 + K;
            float v = (d < DD) ? fminf(fmaxf(row[d], 0.f), 1.f) : 0.f;
            a[j] = (_Float16)v;
        }
        v16h bb;
        #pragma unroll
        for (int j = 0; j < 16; ++j)
            bb[j] = (_Float16)s_ew[d0 + half * 8 + ((j < 8) ? j : (j + 8))];   // zero-padded past d=199
        acc = __builtin_amdgcn_wmma_f32_16x16x32_f16(false, a, false, bb,
                                                     (short)0, acc, false, false);
    }

    // D (16x16 f32) layout: lanes 0-15 hold M=0..7 in acc[0..7]; lanes 16-31 hold M=8..15.
    // All columns identical -> lanes 0-7 emit M=0..7, lanes 16-23 emit M=8..15.
    int mi = -1;
    if (lane < 8)                      mi = lane;
    else if (lane >= 16 && lane < 24)  mi = lane - 8;
    if (mi >= 0) {
        const int vi = (lane < 8) ? lane : (lane - 16);
        float v = acc[0];
        #pragma unroll
        for (int k = 1; k < 8; ++k) if (vi == k) v = acc[k];
        const int so = s_base + mi;
        if (so < SS)
            out[b * SS + so] = 1.0f / (1.0f + __expf(-v));   // sigmoid
    }
}

extern "C" void kernel_launch(void* const* d_in, const int* in_sizes, int n_in,
                              void* d_out, int out_size, void* d_ws, size_t ws_size,
                              hipStream_t stream) {
    const float* cpt  = (const float*)d_in[0];   // concept_x_g1 [NC, D] f32
    const float* xg2  = (const float*)d_in[1];   // x_g2_mini    [B, D]  f32
    const float* w    = (const float*)d_in[2];   // weights      [D, 1]  f32
    const int*   sidx = (const int*)  d_in[3];   // sample_index [B, S]  int
    float* out = (float*)d_out;                  // [B, S] f32

    gather_dot_sigmoid<<<dim3(BB), dim3(416), 0, stream>>>(cpt, xg2, w, sidx, out);
}
